// TropicalLeNet1_50379966382870
// MI455X (gfx1250) — matrix operand order repaired and checked by reading, not compiled, not run on hardware
//
#include <hip/hip_runtime.h>
#include <hip/hip_bf16.h>

typedef __attribute__((ext_vector_type(16))) _Float16 v16h;
typedef __attribute__((ext_vector_type(8)))  _Float16 v8h;
typedef __attribute__((ext_vector_type(8)))  float    v8f;

// ---------------------------------------------------------------------------
// Stage 1: tropical (min-plus) conv 5x5 pad=2 over [256,1,28,28] + 2x2 avgpool
// y[b,o,ph,pw] = 0.25 * sum_{dy,dx} min_{i,j}( xpad[b, 2ph+dy+i-2, 2pw+dx+j-2] + w1[o,i,j] )
// ---------------------------------------------------------------------------
__global__ void conv1_pool_minplus(const float* __restrict__ x,
                                   const float* __restrict__ w1,
                                   float* __restrict__ y) {
    int idx = blockIdx.x * blockDim.x + threadIdx.x;
    const int total = 256 * 6 * 14 * 14;
    if (idx >= total) return;
    int pw = idx % 14;
    int ph = (idx / 14) % 14;
    int o  = (idx / 196) % 6;
    int b  = idx / (196 * 6);

    const float* xb = x + b * 784;
    float wreg[25];
    #pragma unroll
    for (int t = 0; t < 25; ++t) wreg[t] = w1[o * 25 + t];

    float acc = 0.f;
    #pragma unroll
    for (int dy = 0; dy < 2; ++dy) {
        #pragma unroll
        for (int dx = 0; dx < 2; ++dx) {
            int h = 2 * ph + dy, w = 2 * pw + dx;
            float m = 3.4e38f;
            #pragma unroll
            for (int i = 0; i < 5; ++i) {
                int yy = h + i - 2;
                #pragma unroll
                for (int j = 0; j < 5; ++j) {
                    int xx = w + j - 2;
                    float v = (yy >= 0 && yy < 28 && xx >= 0 && xx < 28)
                                  ? xb[yy * 28 + xx] : 0.f;   // zero padding
                    m = fminf(m, v + wreg[i * 5 + j]);
                }
            }
            acc += m;
        }
    }
    y[idx] = acc * 0.25f;   // layout [b][o][ph][pw] matches idx decode
}

// ---------------------------------------------------------------------------
// Stage 2: tropical (max-plus) conv 5x5 pad=0 over [256,6,14,14] + 2x2 avgpool
// Writes flattened f16 activations, padded [256, 416]; cols 400..415 = 0.
// Flatten order: col = o*25 + ph*5 + pw  (matches reshape of [B,16,5,5]).
// ---------------------------------------------------------------------------
__global__ void conv2_pool_maxplus_f16(const float* __restrict__ x1,
                                       const float* __restrict__ w2,
                                       _Float16* __restrict__ xf) {
    int idx = blockIdx.x * blockDim.x + threadIdx.x;
    const int total = 256 * 416;
    if (idx >= total) return;
    int col = idx % 416;
    int b   = idx / 416;
    if (col >= 400) { xf[idx] = (_Float16)0.f; return; }   // K-pad for GEMM1

    int o   = col / 25;
    int rem = col % 25;
    int ph  = rem / 5, pw = rem % 5;

    const float* xb = x1 + b * (6 * 196);
    float acc = 0.f;
    #pragma unroll
    for (int dy = 0; dy < 2; ++dy) {
        #pragma unroll
        for (int dx = 0; dx < 2; ++dx) {
            int h = 2 * ph + dy, w = 2 * pw + dx;        // conv out coord, 0..9
            float s = 0.f;
            for (int c = 0; c < 6; ++c) {
                const float* xc = xb + c * 196;
                const float* wc = w2 + (o * 6 + c) * 25;
                float m = -3.4e38f;
                #pragma unroll
                for (int i = 0; i < 5; ++i) {
                    #pragma unroll
                    for (int j = 0; j < 5; ++j)
                        m = fmaxf(m, xc[(h + i) * 14 + (w + j)] + wc[i * 5 + j]);
                }
                s += m;   // sum over channels of per-channel max
            }
            acc += s;
        }
    }
    xf[idx] = (_Float16)(acc * 0.25f);
}

// ---------------------------------------------------------------------------
// Weight pad + f32 -> f16 convert: src [R,C] -> dst [Rp,Cp], zero outside.
// ---------------------------------------------------------------------------
__global__ void pad_convert_f16(const float* __restrict__ src,
                                _Float16* __restrict__ dst,
                                int R, int C, int Rp, int Cp) {
    int idx = blockIdx.x * blockDim.x + threadIdx.x;
    if (idx >= Rp * Cp) return;
    int c = idx % Cp, r = idx / Cp;
    float v = (r < R && c < C) ? src[r * C + c] : 0.f;
    dst[idx] = (_Float16)v;
}

// ---------------------------------------------------------------------------
// WMMA GEMM: D[m,n] = sum_k A[m,k] * W[n,k]  (+bias, optional ReLU)
//   A   : f16 [256, Kp]  row-major, Kp % 32 == 0, 16B-aligned rows
//   W   : f16 [Np,  Kp]  row-major (this IS B = W^T of the math GEMM)
//   outH: f16 [256, Np]  (feeds next layer; padded cols written as 0), or null
//   outF: f32 [256, NFstride] (final logits), or null
// One wave (32 threads) computes one 16x16 tile via v_wmma_f32_16x16x32_f16.
// grid = (Np/16, 256/16), block = 32.  No divergence before/around the WMMA
// loop, so EXEC is all-ones as the ISA requires.
// ---------------------------------------------------------------------------
__global__ void gemm_wmma_f16(const _Float16* __restrict__ A,
                              const _Float16* __restrict__ W,
                              const float* __restrict__ bias,
                              _Float16* __restrict__ outH,
                              float* __restrict__ outF,
                              int Kp, int Np, int Nreal, int NFstride, int relu) {
    const int lane = threadIdx.x;        // 0..31
    const int n0   = blockIdx.x * 16;
    const int m0   = blockIdx.y * 16;
    const int half = lane >> 4;          // 0: lanes 0-15, 1: lanes 16-31
    const int l15  = lane & 15;

    // A lane base: row m0+l15; K chunks {half*8 .. +7} and {16+half*8 .. +7}
    const _Float16* arow = A + (size_t)(m0 + l15) * Kp + half * 8;
    // B lane base: col n0+l15; K chunks {half*8 .. +7} and {16+half*8 .. +7}
    const _Float16* brow = W + (size_t)(n0 + l15) * Kp + half * 8;

    v8f c = {};
    for (int k0 = 0; k0 < Kp; k0 += 32) {
        v8h alo = *(const v8h*)(arow + k0);        // K = k0 + half*8 + [0..7]
        v8h ahi = *(const v8h*)(arow + k0 + 16);   // K = k0 + 16 + half*8 + [0..7]
        v16h a;
        #pragma unroll
        for (int e = 0; e < 8; ++e) { a[e] = alo[e]; a[e + 8] = ahi[e]; }
        v8h blo = *(const v8h*)(brow + k0);
        v8h bhi = *(const v8h*)(brow + k0 + 16);
        v16h bm;
        #pragma unroll
        for (int e = 0; e < 8; ++e) { bm[e] = blo[e]; bm[e + 8] = bhi[e]; }
        c = __builtin_amdgcn_wmma_f32_16x16x32_f16(
                false, a, false, bm, (short)0, c, false, false);
    }

    const int n = n0 + l15;
    #pragma unroll
    for (int r = 0; r < 8; ++r) {
        int m = m0 + r + half * 8;                 // C/D layout: VGPR r <-> row
        float v = c[r];
        if (n < Nreal) v += bias[n];
        if (relu) v = fmaxf(v, 0.f);
        if (outH) outH[(size_t)m * Np + n] = (_Float16)((n < Nreal) ? v : 0.f);
        if (outF && n < Nreal) outF[(size_t)m * NFstride + n] = v;
    }
}

// ---------------------------------------------------------------------------
// Workspace layout (bytes, 256B aligned regions), total ~1.67 MB:
//   x1  f32 [256,6,14,14]   @ 0          (1,204,224)
//   Xh  f16 [256,416]       @ 1,204,224  (  212,992)
//   W1h f16 [128,416]       @ 1,417,216  (  106,496)
//   Y1  f16 [256,128]       @ 1,523,712  (   65,536)
//   W2h f16 [ 96,128]       @ 1,589,248  (   24,576)
//   Y2  f16 [256, 96]       @ 1,613,824  (   49,152)
//   W3h f16 [ 16, 96]       @ 1,662,976  (    3,072)
// ---------------------------------------------------------------------------
extern "C" void kernel_launch(void* const* d_in, const int* in_sizes, int n_in,
                              void* d_out, int out_size, void* d_ws, size_t ws_size,
                              hipStream_t stream) {
    const float* x   = (const float*)d_in[0];
    const float* w1  = (const float*)d_in[1];
    const float* w2  = (const float*)d_in[2];
    const float* fw1 = (const float*)d_in[3];
    const float* fb1 = (const float*)d_in[4];
    const float* fw2 = (const float*)d_in[5];
    const float* fb2 = (const float*)d_in[6];
    const float* fw3 = (const float*)d_in[7];
    const float* fb3 = (const float*)d_in[8];
    float* out = (float*)d_out;

    char* ws = (char*)d_ws;
    float*    x1  = (float*)   (ws + 0);
    _Float16* Xh  = (_Float16*)(ws + 1204224);
    _Float16* W1h = (_Float16*)(ws + 1417216);
    _Float16* Y1  = (_Float16*)(ws + 1523712);
    _Float16* W2h = (_Float16*)(ws + 1589248);
    _Float16* Y2  = (_Float16*)(ws + 1613824);
    _Float16* W3h = (_Float16*)(ws + 1662976);

    conv1_pool_minplus<<<(256 * 6 * 196 + 255) / 256, 256, 0, stream>>>(x, w1, x1);

    pad_convert_f16<<<(128 * 416 + 255) / 256, 256, 0, stream>>>(fw1, W1h, 120, 400, 128, 416);
    pad_convert_f16<<<( 96 * 128 + 255) / 256, 256, 0, stream>>>(fw2, W2h,  84, 120,  96, 128);
    pad_convert_f16<<<( 16 *  96 + 255) / 256, 256, 0, stream>>>(fw3, W3h,  10,  84,  16,  96);

    conv2_pool_maxplus_f16<<<(256 * 416 + 255) / 256, 256, 0, stream>>>(x1, w2, Xh);

    dim3 g1(8, 16), g2(6, 16), g3(1, 16);
    gemm_wmma_f16<<<g1, 32, 0, stream>>>(Xh, W1h, fb1, Y1, nullptr, 416, 128, 120,  0, 1);
    gemm_wmma_f16<<<g2, 32, 0, stream>>>(Y1, W2h, fb2, Y2, nullptr, 128,  96,  84,  0, 1);
    gemm_wmma_f16<<<g3, 32, 0, stream>>>(Y2, W3h, fb3, nullptr, out,  96,  16,  10, 10, 0);
}
